// Parent_LSTM_38689065402499
// MI455X (gfx1250) — matrix operand order repaired and checked by reading, not compiled, not run on hardware
//
#include <hip/hip_runtime.h>
#include <stdint.h>

#define BATCH    4096
#define WORD_DIM 1024
#define TAG_DIM  128
#define MEM_DIM  2048
#define KDIM     (WORD_DIM + TAG_DIM + MEM_DIM)   // 3200
#define NDIM     (4 * MEM_DIM)                    // 8192
#define K4       (KDIM / 4)                       // 800

#define MTILE 128
#define KT    64                      // k-step: two x32 WMMA sub-steps per tile
#define RS    144                     // LDS row stride bytes: 64 bf16 =128B +16B pad (36-bank stride, conflict-free)
#define ABUF  (128 * RS)              // 18432 B per A tile
#define BUFSZ (2 * ABUF)              // A + B per stage = 36864 B; double buffered = 73728 B

typedef __attribute__((ext_vector_type(16))) __bf16         v16bf;
typedef __attribute__((ext_vector_type(8)))  float          v8f;
typedef __attribute__((ext_vector_type(4)))  unsigned int   u32x4;
typedef __attribute__((ext_vector_type(4)))  float          f32x4;
typedef __attribute__((ext_vector_type(4)))  unsigned short u16x4;

__device__ __forceinline__ uint16_t f2bf(float f) {
    uint32_t u = __float_as_uint(f);
    u += 0x7FFFu + ((u >> 16) & 1u);          // round-to-nearest-even
    return (uint16_t)(u >> 16);
}

__device__ __forceinline__ float sigmoidf_fast(float x) {
    return 1.0f / (1.0f + __expf(-x));
}

// CDNA5 async copy: global -> LDS direct, tracked by ASYNCcnt (no VGPR staging).
// VDST VGPR holds the per-lane LDS byte address (= low 32 bits of generic shared ptr).
__device__ __forceinline__ void async_copy_b128(uint32_t lds_addr, const void* gptr) {
    asm volatile("global_load_async_to_lds_b128 %0, %1, off"
                 :: "v"(lds_addr), "v"(gptr)
                 : "memory");
}
#define WAIT_ASYNC(n) asm volatile("s_wait_asynccnt " #n ::: "memory")

// ---------------------------------------------------------------------------
// Pack X = [word | tag | h_prev] as bf16 [BATCH x KDIM]
// ---------------------------------------------------------------------------
__global__ __launch_bounds__(256) void pack_x_kernel(
    const float* __restrict__ word, const float* __restrict__ tag,
    const float* __restrict__ hp, uint16_t* __restrict__ Xb)
{
    const int nchunk = BATCH * K4;
    const int stride = gridDim.x * 256;
    for (int idx = blockIdx.x * 256 + threadIdx.x; idx < nchunk; idx += stride) {
        int b = idx / K4;
        int k = (idx - b * K4) * 4;   // chunks never straddle source boundaries (1024,1152 % 4 == 0)
        const float* src;
        if (k < WORD_DIM)                src = word + (size_t)b * WORD_DIM + k;
        else if (k < WORD_DIM + TAG_DIM) src = tag + (size_t)b * TAG_DIM + (k - WORD_DIM);
        else                             src = hp  + (size_t)b * MEM_DIM + (k - WORD_DIM - TAG_DIM);
        f32x4 v = *(const f32x4*)src;
        u16x4 o;
        o.x = f2bf(v.x); o.y = f2bf(v.y); o.z = f2bf(v.z); o.w = f2bf(v.w);
        *(u16x4*)(Xb + (size_t)idx * 4) = o;
    }
}

// ---------------------------------------------------------------------------
// Pack W = [Ww | Wt | Wh] as bf16 [NDIM x KDIM], row r = g*2048 + m
// ---------------------------------------------------------------------------
__global__ __launch_bounds__(256) void pack_w_kernel(
    const float* __restrict__ Ww, const float* __restrict__ Wt,
    const float* __restrict__ Wh, uint16_t* __restrict__ Wb)
{
    const int nchunk = NDIM * K4;
    const int stride = gridDim.x * 256;
    for (int idx = blockIdx.x * 256 + threadIdx.x; idx < nchunk; idx += stride) {
        int r = idx / K4;                 // r = g*2048 + m matches Ww/Wt/Wh row layout directly
        int k = (idx - r * K4) * 4;
        const float* src;
        if (k < WORD_DIM)                src = Ww + (size_t)r * WORD_DIM + k;
        else if (k < WORD_DIM + TAG_DIM) src = Wt + (size_t)r * TAG_DIM + (k - WORD_DIM);
        else                             src = Wh + (size_t)r * MEM_DIM + (k - WORD_DIM - TAG_DIM);
        f32x4 v = *(const f32x4*)src;
        u16x4 o;
        o.x = f2bf(v.x); o.y = f2bf(v.y); o.z = f2bf(v.z); o.w = f2bf(v.w);
        *(u16x4*)(Wb + (size_t)idx * 4) = o;
    }
}

__global__ __launch_bounds__(256) void pack_bias_kernel(
    const float* __restrict__ bw, const float* __restrict__ bt,
    const float* __restrict__ bh, float* __restrict__ bias)
{
    int n = blockIdx.x * 256 + threadIdx.x;
    if (n < NDIM) bias[n] = bw[n] + bt[n] + bh[n];
}

// ---------------------------------------------------------------------------
// Fused GEMM + LSTM epilogue, double-buffered via async global->LDS copies,
// with register-level two-deep pipelining of B fragments.
// Block = 8 wave32s. Tile: 128 batch rows x (4 gates x 32 mem cols).
// ---------------------------------------------------------------------------
__global__ __launch_bounds__(256) void lstm_gemm_kernel(
    const uint16_t* __restrict__ Xb, const uint16_t* __restrict__ Wb,
    const float* __restrict__ bias, const float* __restrict__ c_prev,
    float* __restrict__ h_out, float* __restrict__ c_out)
{
    __shared__ __align__(16) uint8_t lds[2 * BUFSZ];   // 73.7 KB: [buf][A|B]

    const int tid   = threadIdx.x;
    const int wave  = tid >> 5;
    const int lane  = tid & 31;
    const int l16   = lane & 15;
    const int lhalf = lane >> 4;

    const int m_base = blockIdx.x * MTILE;   // batch tile
    const int mem0   = blockIdx.y * 32;      // mem-column tile (shared by all 4 gates)

    // --- copy assignment: per tile, each matrix = 128 rows x 128 B = 1024 chunks of 16 B,
    // 4 chunks per thread per matrix. chunk s: row = s>>3, j = s&7.
    uint32_t ldsOffA[4], ldsOffB[4];
    const uint16_t* gA[4];
    const uint16_t* gB[4];
#pragma unroll
    for (int q = 0; q < 4; ++q) {
        const int s = tid + 256 * q;
        const int row = s >> 3, j = s & 7;
        ldsOffA[q] = (uint32_t)(row * RS + j * 16);
        ldsOffB[q] = (uint32_t)(ABUF + row * RS + j * 16);
        gA[q] = Xb + (size_t)(m_base + row) * KDIM + j * 8;
        const int wrow = (row >> 5) * MEM_DIM + (mem0 + (row & 31));   // gate*2048 + mcol
        gB[q] = Wb + (size_t)wrow * KDIM + j * 8;
    }
    const uint32_t ldsBase = (uint32_t)(uintptr_t)(&lds[0]);   // low 32 bits = LDS byte address

    v8f acc[8];
#pragma unroll
    for (int t = 0; t < 8; ++t) acc[t] = (v8f)(0.0f);

    // prologue: issue tile 0 into buffer 0 (8 async ops per wave)
#pragma unroll
    for (int q = 0; q < 4; ++q) {
        async_copy_b128(ldsBase + ldsOffA[q], gA[q]);
        async_copy_b128(ldsBase + ldsOffB[q], gB[q]);
    }

    union Frag { u32x4 u[2]; v16bf v; };

    const int NITER = KDIM / KT;   // 50
#pragma unroll 1
    for (int i = 0; i < NITER; ++i) {
        const int cur = i & 1;

        // issue next tile into the other buffer (its last readers finished at the
        // tail barrier of iteration i-1, which precedes this point in program order)
        if (i + 1 < NITER) {
            const uint32_t nb = ldsBase + (uint32_t)((1 - cur) * BUFSZ);
            const int kn = (i + 1) * KT;
#pragma unroll
            for (int q = 0; q < 4; ++q) {
                async_copy_b128(nb + ldsOffA[q], gA[q] + kn);
                async_copy_b128(nb + ldsOffB[q], gB[q] + kn);
            }
            WAIT_ASYNC(8);   // copies complete in order: <=8 outstanding => tile i resident
        } else {
            WAIT_ASYNC(0);
        }
        __syncthreads();      // publish tile i across all waves

        const uint8_t* aRow = lds + cur * BUFSZ + (size_t)(16 * wave + l16) * RS;
        const uint8_t* bBuf = lds + cur * BUFSZ + ABUF;
        const uint8_t* bLane = bBuf + (size_t)l16 * RS + lhalf * 32;  // t=0 column base

        Frag af[2], bf[2];
        // preload A(ks=0) and B(ks=0,t=0)
        af[0].u[0] = *(const u32x4*)(aRow + lhalf * 32);              // K = 16h..16h+7
        af[0].u[1] = *(const u32x4*)(aRow + 16 + lhalf * 32);         // K = 16h+8..16h+15
        bf[0].u[0] = *(const u32x4*)(bLane);
        bf[0].u[1] = *(const u32x4*)(bLane + 16);

#pragma unroll
        for (int ks = 0; ks < 2; ++ks) {   // two x32 sub-steps of the 64-wide k tile
            if (ks == 0) {                 // prefetch second A fragment early
                af[1].u[0] = *(const u32x4*)(aRow + 64 + lhalf * 32);
                af[1].u[1] = *(const u32x4*)(aRow + 80 + lhalf * 32);
            }
#pragma unroll
            for (int t = 0; t < 8; ++t) {
                // two-deep register pipeline: fetch B for the *next* WMMA before
                // issuing the current one, so the pre-WMMA wait is dscnt<=2, not 0
                if (t + 1 < 8) {
                    const uint8_t* bRow = bBuf + (size_t)((t + 1) * 16 + l16) * RS
                                        + ks * 64 + lhalf * 32;
                    bf[(t + 1) & 1].u[0] = *(const u32x4*)(bRow);
                    bf[(t + 1) & 1].u[1] = *(const u32x4*)(bRow + 16);
                } else if (ks == 0) {      // (ks=0,t=7) prefetches (ks=1,t=0) into bf[0]
                    bf[0].u[0] = *(const u32x4*)(bLane + 64);
                    bf[0].u[1] = *(const u32x4*)(bLane + 64 + 16);
                }
                acc[t] = __builtin_amdgcn_wmma_f32_16x16x32_bf16(
                    false, af[ks].v, false, bf[t & 1].v, (short)0, acc[t], false, false);
            }
        }
        __syncthreads();      // all waves done reading buf[cur] before it is refilled
    }

    // ------------------ fused LSTM epilogue ------------------
    // N-subtiles: t = gate*2 + tc ; acc element v <-> M = v + 8*lhalf, N = l16
#pragma unroll
    for (int tc = 0; tc < 2; ++tc) {
        const int mcol = mem0 + 16 * tc + l16;
        const float bi  = bias[0 * MEM_DIM + mcol];
        const float bff = bias[1 * MEM_DIM + mcol];
        const float bo  = bias[2 * MEM_DIM + mcol];
        const float bu  = bias[3 * MEM_DIM + mcol];
        const v8f zi = acc[0 + tc];
        const v8f zf = acc[2 + tc];
        const v8f zo = acc[4 + tc];
        const v8f zu = acc[6 + tc];
#pragma unroll
        for (int v = 0; v < 8; ++v) {
            const int brow = m_base + 16 * wave + 8 * lhalf + v;
            const size_t o = (size_t)brow * MEM_DIM + mcol;
            const float ig = sigmoidf_fast(zi[v] + bi);
            const float fg = sigmoidf_fast(zf[v] + bff);
            const float og = sigmoidf_fast(zo[v] + bo);
            const float ug = tanhf(zu[v] + bu);
            const float c  = ig * ug + fg * c_prev[o];
            const float h  = og * tanhf(c);
            h_out[o] = h;
            c_out[o] = c;
        }
    }
}

// ---------------------------------------------------------------------------
extern "C" void kernel_launch(void* const* d_in, const int* in_sizes, int n_in,
                              void* d_out, int out_size, void* d_ws, size_t ws_size,
                              hipStream_t stream)
{
    const float* word  = (const float*)d_in[0];
    const float* tag   = (const float*)d_in[1];
    const float* hprev = (const float*)d_in[2];
    const float* cprev = (const float*)d_in[3];
    const float* Ww    = (const float*)d_in[4];
    const float* bw    = (const float*)d_in[5];
    const float* Wt    = (const float*)d_in[6];
    const float* bt    = (const float*)d_in[7];
    const float* Wh    = (const float*)d_in[8];
    const float* bh    = (const float*)d_in[9];

    // workspace layout (16B aligned): Xb (26.2MB) | Wb (52.4MB) | bias (32KB)
    uint16_t* Xb   = (uint16_t*)d_ws;
    uint16_t* Wb   = (uint16_t*)((char*)d_ws + (size_t)BATCH * KDIM * 2);
    float*    bias = (float*)((char*)d_ws + (size_t)BATCH * KDIM * 2 + (size_t)NDIM * KDIM * 2);

    float* h_out = (float*)d_out;
    float* c_out = h_out + (size_t)BATCH * MEM_DIM;

    pack_x_kernel<<<4096, 256, 0, stream>>>(word, tag, hprev, Xb);
    pack_w_kernel<<<8192, 256, 0, stream>>>(Ww, Wt, Wh, Wb);
    pack_bias_kernel<<<NDIM / 256, 256, 0, stream>>>(bw, bt, bh, bias);

    dim3 grid(BATCH / MTILE, MEM_DIM / 32);   // 32 x 64 = 2048 blocks
    lstm_gemm_kernel<<<grid, 256, 0, stream>>>(Xb, Wb, bias, cprev, h_out, c_out);
}
